// IG_MSA_18442589569203
// MI455X (gfx1250) — matrix operand order repaired and checked by reading, not compiled, not run on hardware
//
#include <hip/hip_runtime.h>
#include <hip/hip_bf16.h>
#include <math.h>

typedef __bf16 v16bf __attribute__((ext_vector_type(16)));
typedef float  v8f   __attribute__((ext_vector_type(8)));
typedef unsigned int u32x4 __attribute__((ext_vector_type(4)));
typedef int          i32x8 __attribute__((ext_vector_type(8)));
typedef int          i32x4 __attribute__((ext_vector_type(4)));

#define P_PIX 16384   // H*W = 128*128
#define BATCH 8
#define HEADS 8

// ---------------------------------------------------------------------------
// GEMM: Y[b, oc, p] = sum_ic W[oc, ic] * X[b, ic, p]   (bf16 WMMA, f32 accum)
// grid: x = P/64 pixel tiles, y = Cout/128, z = batch. block = 256 (8 waves).
// X tiles [32 x 64] are DMA'd into double-buffered LDS by the Tensor Data
// Mover (tensor_load_to_lds, TENSORcnt), overlapping with WMMA compute.
// ---------------------------------------------------------------------------
__global__ __launch_bounds__(256)
void gemm1x1_bf16_kernel(const float* __restrict__ X, const float* __restrict__ W,
                         float* __restrict__ Y, int Cin, int Cout) {
    const int P = P_PIX;
    __shared__ float xs[2][32][64];  // double-buffered K-tile (TDM writes rows densely)
    const int tid  = threadIdx.x;
    const int wid  = tid >> 5;
    const int lane = tid & 31;
    const int g    = lane >> 4;   // half-wave group
    const int m    = lane & 15;
    const int b    = blockIdx.z;
    const int oc0  = blockIdx.y * 128 + wid * 16;
    const int px0  = blockIdx.x * 64;
    const float* Xt = X + (size_t)b * Cin * P + px0;   // tile column base
    const int nk = Cin / 32;

    // LDS byte offset of xs: flat LDS-aperture pointers carry the LDS offset
    // in their low 32 bits (ISA 10.2 per-aperture address calculation).
    const unsigned lds_base = (unsigned)(size_t)(void*)(&xs[0][0][0]);

    // Issue a TDM load of X[k0..k0+32) x [px0..px0+64) into buffer (ki&1).
    auto issue_tile = [&](int ki) {
        const unsigned long long gaddr =
            (unsigned long long)(size_t)(Xt + (size_t)ki * 32 * P);
        const unsigned laddr = lds_base + (unsigned)(ki & 1) * (32u * 64u * 4u);
        u32x4 g0;
        g0.x = 1u;                                   // count=1, user descriptor
        g0.y = laddr;                                // lds_addr [63:32]
        g0.z = (unsigned)(gaddr & 0xFFFFFFFFu);      // global_addr [95:64]
        g0.w = (unsigned)((gaddr >> 32) & 0x1FFFFFFu) | (2u << 30);  // addr hi | type=2
        i32x8 g1;
        g1[0] = 0x00020000;            // data_size=2 (4B), no flags
        g1[1] = 0x40000000;            // tensor_dim0[15:0]=16384 @ bits 63:48
        g1[2] = (int)(32u << 16);      // tensor_dim1[15:0]=32 @ bits 95:80
        g1[3] = (int)(64u << 16);      // tile_dim0=64 @ bits 127:112
        g1[4] = 32;                    // tile_dim1=32 @ bits 143:128 (tile_dim2=0)
        g1[5] = 16384;                 // tensor_dim0_stride[31:0] @ bits 191:160
        g1[6] = 0;                     // stride hi / dim1_stride lo
        g1[7] = 0;
        i32x4 z4 = {0, 0, 0, 0};
        i32x8 z8 = {0, 0, 0, 0, 0, 0, 0, 0};
        __builtin_amdgcn_tensor_load_to_lds(g0, g1, z4, z4, z8, 0);
    };

    if (wid == 0) issue_tile(0);

    v8f acc[4] = {};

    for (int ki = 0; ki < nk; ++ki) {
        __builtin_amdgcn_s_wait_tensorcnt(0);  // tile ki landed (no-op for waves 1..7)
        __syncthreads();
        if (wid == 0 && ki + 1 < nk) issue_tile(ki + 1);  // prefetch next tile

        const int buf = ki & 1;
        const int k0  = ki * 32;

        // A fragment: W[oc0+m][k0 + K(e,g)], 16-bit A layout (16x32)
        v16bf a;
        {
            const float* wrow = W + (size_t)(oc0 + m) * Cin + k0;
            #pragma unroll
            for (int e = 0; e < 16; ++e) a[e] = (__bf16)wrow[g * 16 + e];
        }

        // B fragments (32x16 slices of the LDS tile) + WMMA
        #pragma unroll
        for (int s = 0; s < 4; ++s) {
            v16bf bf;
            #pragma unroll
            for (int e = 0; e < 16; ++e) bf[e] = (__bf16)xs[buf][g * 16 + e][s * 16 + m];
            acc[s] = __builtin_amdgcn_wmma_f32_16x16x32_bf16(
                false, a, false, bf, (short)0, acc[s], false, false);
        }
        __syncthreads();  // all reads of buf done before TDM reuses it
    }

    // store: C layout -> row oc0 + r + 8*g, col px0 + s*16 + m
    #pragma unroll
    for (int s = 0; s < 4; ++s) {
        #pragma unroll
        for (int r = 0; r < 8; ++r) {
            const int oc = oc0 + r + 8 * g;
            Y[((size_t)b * Cout + oc) * P + px0 + s * 16 + m] = acc[s][r];
        }
    }
}

// ---------------------------------------------------------------------------
// Depthwise 3x3 conv, SAME padding; optional exact GELU, optional residual add.
// ---------------------------------------------------------------------------
__global__ __launch_bounds__(256)
void dwconv3x3_kernel(const float* __restrict__ in, const float* __restrict__ w,
                      float* __restrict__ out, const float* __restrict__ residual,
                      int C, int apply_gelu) {
    const int Hh = 128, Ww = 128;
    const size_t idx = (size_t)blockIdx.x * blockDim.x + threadIdx.x;
    const size_t total = (size_t)BATCH * C * Hh * Ww;
    if (idx >= total) return;
    const int x = (int)(idx & (Ww - 1));
    const int y = (int)((idx >> 7) & (Hh - 1));
    const size_t bc = idx >> 14;              // b*C + c
    const int c = (int)(bc % (size_t)C);
    const float* wp = w + (size_t)c * 9;
    const float* base = in + (bc << 14);
    float s = 0.f;
    #pragma unroll
    for (int dy = -1; dy <= 1; ++dy) {
        const int yy = y + dy;
        if (yy < 0 || yy >= Hh) continue;
        #pragma unroll
        for (int dx = -1; dx <= 1; ++dx) {
            const int xx = x + dx;
            if (xx < 0 || xx >= Ww) continue;
            s += wp[(dy + 1) * 3 + (dx + 1)] * base[yy * Ww + xx];
        }
    }
    if (apply_gelu) s = 0.5f * s * (1.f + erff(s * 0.70710678118f));
    if (residual) s += residual[idx];
    out[idx] = s;
}

// ---------------------------------------------------------------------------
// Per-(b, channel) inverse L2 norm over N=16384 for q (ch 0..255) and k (256..511)
// of qkv_dw. One block per row; LDS tree reduction.
// ---------------------------------------------------------------------------
__global__ __launch_bounds__(256)
void l2norm_scale_kernel(const float* __restrict__ qkv, float* __restrict__ scales) {
    const int b = blockIdx.x >> 9;
    const int c = blockIdx.x & 511;
    const float* row = qkv + ((size_t)b * 768 + c) * P_PIX;
    float s = 0.f;
    for (int i = threadIdx.x; i < P_PIX; i += 256) {
        const float v = row[i];
        s += v * v;
    }
    __shared__ float red[256];
    red[threadIdx.x] = s;
    __syncthreads();
    #pragma unroll
    for (int off = 128; off > 0; off >>= 1) {
        if (threadIdx.x < off) red[threadIdx.x] += red[threadIdx.x + off];
        __syncthreads();
    }
    if (threadIdx.x == 0)
        scales[blockIdx.x] = 1.0f / fmaxf(sqrtf(red[0]), 1e-12f);
}

// ---------------------------------------------------------------------------
// attn[b,h] = softmax( (q @ k^T) * s_q s_k * temperature )  -- 32x32, K=16384.
// One block per (b,head); 8 waves split K (2048 each); partial C tiles reduced
// via shared-memory atomics; softmax by first 32 threads.
// ---------------------------------------------------------------------------
__global__ __launch_bounds__(256)
void attn_kernel(const float* __restrict__ qkv, const float* __restrict__ scales,
                 const float* __restrict__ temperature, float* __restrict__ attn_out) {
    const int P = P_PIX;
    const int b = blockIdx.x >> 3, head = blockIdx.x & 7;
    const int tid = threadIdx.x;
    const int wid = tid >> 5, lane = tid & 31, g = lane >> 4, m = lane & 15;
    const float* q = qkv + ((size_t)b * 768 + head * 32) * P;
    const float* k = qkv + ((size_t)b * 768 + 256 + head * 32) * P;

    __shared__ float sattn[32][33];
    for (int i = tid; i < 32 * 33; i += 256) (&sattn[0][0])[i] = 0.f;
    __syncthreads();

    v8f acc[2][2] = {};
    const int n0base = wid * 2048;
    for (int step = 0; step < 64; ++step) {
        const int n0 = n0base + step * 32;
        v16bf afrag[2], bfrag[2];
        #pragma unroll
        for (int mi = 0; mi < 2; ++mi) {
            const float* qr = q + (size_t)(mi * 16 + m) * P + n0;
            #pragma unroll
            for (int e = 0; e < 16; ++e) afrag[mi][e] = (__bf16)qr[g * 16 + e];
        }
        #pragma unroll
        for (int ni = 0; ni < 2; ++ni) {
            // B[K=n][N=d] = k[d][n]; lane holds column d=m, contiguous 16 n-values
            const float* kr = k + (size_t)(ni * 16 + m) * P + n0 + g * 16;
            #pragma unroll
            for (int e = 0; e < 16; ++e) bfrag[ni][e] = (__bf16)kr[e];
        }
        #pragma unroll
        for (int mi = 0; mi < 2; ++mi)
            #pragma unroll
            for (int ni = 0; ni < 2; ++ni)
                acc[mi][ni] = __builtin_amdgcn_wmma_f32_16x16x32_bf16(
                    false, afrag[mi], false, bfrag[ni], (short)0, acc[mi][ni], false, false);
    }
    #pragma unroll
    for (int mi = 0; mi < 2; ++mi)
        #pragma unroll
        for (int ni = 0; ni < 2; ++ni)
            #pragma unroll
            for (int r = 0; r < 8; ++r)
                atomicAdd(&sattn[mi * 16 + r + 8 * g][ni * 16 + m], acc[mi][ni][r]);
    __syncthreads();

    if (tid < 32) {
        const int c = tid;
        const float sq = scales[b * 512 + head * 32 + c];
        const float temp = temperature[head];
        float vals[32];
        float mx = -1e30f;
        #pragma unroll
        for (int d = 0; d < 32; ++d) {
            const float sk = scales[b * 512 + 256 + head * 32 + d];
            const float v = sattn[c][d] * sq * sk * temp;
            vals[d] = v;
            mx = fmaxf(mx, v);
        }
        float sum = 0.f;
        #pragma unroll
        for (int d = 0; d < 32; ++d) { vals[d] = expf(vals[d] - mx); sum += vals[d]; }
        const float inv = 1.f / sum;
        float* dst = attn_out + (size_t)blockIdx.x * 1024 + c * 32;
        #pragma unroll
        for (int d = 0; d < 32; ++d) dst[d] = vals[d] * inv;
    }
}

// ---------------------------------------------------------------------------
// out[b, head*32+c, p] = attn[b,h] @ (v * illu)   -- K=32: ONE wmma per tile.
// grid: x = 64 px-blocks of 256, y = b*8+head. block = 256 (8 waves x 32 px).
// v1 tile staged in LDS with coalesced loads + fused illumination multiply.
// ---------------------------------------------------------------------------
__global__ __launch_bounds__(256)
void av_kernel(const float* __restrict__ qkv, const float* __restrict__ illu,
               const float* __restrict__ attn, float* __restrict__ out) {
    const int P = P_PIX;
    const int b = blockIdx.y >> 3, head = blockIdx.y & 7;
    const int pxBlock = blockIdx.x * 256;
    const int tid = threadIdx.x, wid = tid >> 5, lane = tid & 31;
    const int g = lane >> 4, m = lane & 15;
    const float* vbase = qkv + ((size_t)b * 768 + 512 + head * 32) * P;
    const float* ibase = illu + ((size_t)b * 256 + head * 32) * P;

    __shared__ float vs[32][260];
    {
        const int row  = tid >> 3;         // 0..31 (k index)
        const int col0 = (tid & 7) * 32;   // 0..224
        const float* vr = vbase + (size_t)row * P + pxBlock + col0;
        const float* ir = ibase + (size_t)row * P + pxBlock + col0;
        #pragma unroll
        for (int j = 0; j < 32; ++j) vs[row][col0 + j] = vr[j] * ir[j];
    }
    __syncthreads();

    // A fragments from attn (32x32 f32 -> bf16), 2 row tiles
    const float* am = attn + (size_t)blockIdx.y * 1024;
    v16bf afrag[2];
    #pragma unroll
    for (int mi = 0; mi < 2; ++mi) {
        const float* ar = am + (size_t)(mi * 16 + m) * 32;
        #pragma unroll
        for (int e = 0; e < 16; ++e) afrag[mi][e] = (__bf16)ar[g * 16 + e];
    }

    v8f acc[2][2] = {};
    #pragma unroll
    for (int s = 0; s < 2; ++s) {
        const int pc = wid * 32 + s * 16;
        v16bf bfrag;
        #pragma unroll
        for (int e = 0; e < 16; ++e) bfrag[e] = (__bf16)vs[g * 16 + e][pc + m];
        #pragma unroll
        for (int mi = 0; mi < 2; ++mi)
            acc[mi][s] = __builtin_amdgcn_wmma_f32_16x16x32_bf16(
                false, afrag[mi], false, bfrag, (short)0, acc[mi][s], false, false);
    }

    #pragma unroll
    for (int mi = 0; mi < 2; ++mi)
        #pragma unroll
        for (int s = 0; s < 2; ++s)
            #pragma unroll
            for (int r = 0; r < 8; ++r) {
                const int c = head * 32 + mi * 16 + r + 8 * g;
                out[((size_t)b * 256 + c) * P + pxBlock + wid * 32 + s * 16 + m] = acc[mi][s][r];
            }
}

// ---------------------------------------------------------------------------
extern "C" void kernel_launch(void* const* d_in, const int* in_sizes, int n_in,
                              void* d_out, int out_size, void* d_ws, size_t ws_size,
                              hipStream_t stream) {
    const float* x_in     = (const float*)d_in[0];
    const float* illu_fea = (const float*)d_in[1];
    const float* qkv_w    = (const float*)d_in[2];
    const float* qkv_dw_w = (const float*)d_in[3];
    const float* proj_w   = (const float*)d_in[4];
    const float* pos1_w   = (const float*)d_in[5];
    const float* pos2_w   = (const float*)d_in[6];
    const float* temp     = (const float*)d_in[7];
    float* out = (float*)d_out;

    const size_t P = P_PIX;
    const size_t nQkv = (size_t)BATCH * 768 * P;   // elements
    const size_t nMap = (size_t)BATCH * 256 * P;

    float* qkv_lin = (float*)d_ws;                 // [8,768,P]  (reused below)
    float* qkv_dw  = qkv_lin + nQkv;               // [8,768,P]
    float* scales  = qkv_dw + nQkv;                // [8,512]
    float* attn    = scales + (size_t)BATCH * 512; // [64,32,32]
    // reuse qkv_lin region after dwconv consumed it:
    float* attn_out = qkv_lin;                     // [8,256,P]
    float* proj_out = qkv_lin + nMap;              // [8,256,P]
    float* pos_tmp  = qkv_lin + 2 * nMap;          // [8,256,P]

    // 1) qkv = conv1x1(x_in)
    gemm1x1_bf16_kernel<<<dim3(P / 64, 6, BATCH), 256, 0, stream>>>(
        x_in, qkv_w, qkv_lin, 256, 768);
    // 2) qkv = dwconv3x3(qkv)
    dwconv3x3_kernel<<<(unsigned)(nQkv / 256), 256, 0, stream>>>(
        qkv_lin, qkv_dw_w, qkv_dw, nullptr, 768, 0);
    // 3) inverse L2 norms for q, k rows
    l2norm_scale_kernel<<<BATCH * 512, 256, 0, stream>>>(qkv_dw, scales);
    // 4) attn = softmax(qk^T * scales * temperature)
    attn_kernel<<<BATCH * HEADS, 256, 0, stream>>>(qkv_dw, scales, temp, attn);
    // 5) out = attn @ (v * illu)
    av_kernel<<<dim3(P / 256, BATCH * HEADS), 256, 0, stream>>>(
        qkv_dw, illu_fea, attn, attn_out);
    // 6) proj conv1x1
    gemm1x1_bf16_kernel<<<dim3(P / 64, 2, BATCH), 256, 0, stream>>>(
        attn_out, proj_w, proj_out, 256, 256);
    // 7) pos = dwconv -> gelu
    dwconv3x3_kernel<<<(unsigned)(nMap / 256), 256, 0, stream>>>(
        proj_out, pos1_w, pos_tmp, nullptr, 256, 1);
    // 8) final = proj_out + dwconv(pos)
    dwconv3x3_kernel<<<(unsigned)(nMap / 256), 256, 0, stream>>>(
        pos_tmp, pos2_w, out, proj_out, 256, 0);
}
